// JobShopGraphEncoder_41369124995417
// MI455X (gfx1250) — matrix operand order repaired and checked by reading, not compiled, not run on hardware
//
#include <hip/hip_runtime.h>

#define B_   16
#define N_   1024
#define FIN  64
#define H_   256
#define NEGV (-9.0e15f)
#define ALPHA_ 0.2f

typedef __attribute__((ext_vector_type(16))) __bf16 v16bf;
typedef __attribute__((ext_vector_type(8)))  __bf16 v8bf;
typedef __attribute__((ext_vector_type(8)))  float  v8f;
typedef __attribute__((ext_vector_type(4)))  float  v4f;

static __device__ __forceinline__ v16bf cat8(v8bf lo, v8bf hi) {
  return __builtin_shufflevector(lo, hi, 0,1,2,3,4,5,6,7,8,9,10,11,12,13,14,15);
}

// ---------------------------------------------------------------------------
// C[M,256] = act(A[M,K] @ W[K,256] + bias).
// W is PRE-TRANSPOSED bf16: Wt[h][k]. No LDS, no barriers: A and B fragments
// are loaded directly from global (L2-resident) in one shared contiguous-K order (A and B match).
//   A frag (16x32 bf16): lane(m16,hi) = rows m16, 16 contiguous K at 16*hi
//   B frag (32x16 bf16): lane(n,hi)   = 16 contiguous K halves at 16*hi of Wt row n
// Block: 256 thr (8 waves); wave w -> channels [32w,32w+32) (two D tiles).
// ---------------------------------------------------------------------------
template<bool ABF16>
__global__ void __launch_bounds__(256)
gemm_wmma_kernel(const void* __restrict__ Ain, const __bf16* __restrict__ Wt,
                 const float* __restrict__ bias,
                 float* __restrict__ outF, __bf16* __restrict__ outBf,
                 int K, int relu)
{
  const int tid  = threadIdx.x;
  const int lane = tid & 31, wave = tid >> 5;
  const int hi   = lane >> 4, m16 = lane & 15;
  const int i0   = blockIdx.x * 16;
  const int cb   = wave * 32;
  const int row  = i0 + m16;

  v8f acc0 = {}; v8f acc1 = {};
  const __bf16* wt0 = Wt + (cb + m16) * K + 16 * hi;
  const __bf16* wt1 = Wt + (cb + 16 + m16) * K + 16 * hi;

  for (int kb = 0; kb < K; kb += 32) {
    v16bf afrag;
    if (ABF16) {
      const __bf16* A = (const __bf16*)Ain + row * K + kb + hi * 16;
      v8bf alo = *(const v8bf*)A;
      v8bf ahi = *(const v8bf*)(A + 8);
      afrag = cat8(alo, ahi);
      __builtin_prefetch(A + 32, 0, 1);
    } else {
      const float* A = (const float*)Ain + row * K + kb + hi * 16;
      v8f flo = *(const v8f*)A;
      v8f fhi = *(const v8f*)(A + 8);
      v8bf alo, ahi;
#pragma unroll
      for (int i = 0; i < 8; ++i) { alo[i] = (__bf16)flo[i]; ahi[i] = (__bf16)fhi[i]; }
      afrag = cat8(alo, ahi);
      __builtin_prefetch(A + 32, 0, 1);
    }
    v16bf b0 = *(const v16bf*)(wt0 + kb);
    acc0 = __builtin_amdgcn_wmma_f32_16x16x32_bf16(false, afrag, false, b0,
                                                   (short)0, acc0, false, false);
    v16bf b1 = *(const v16bf*)(wt1 + kb);
    acc1 = __builtin_amdgcn_wmma_f32_16x16x32_bf16(false, afrag, false, b1,
                                                   (short)0, acc1, false, false);
    __builtin_prefetch(wt0 + kb + 32, 0, 1);
  }

#pragma unroll
  for (int t = 0; t < 2; ++t) {
    v8f acc = t ? acc1 : acc0;
    int ch = cb + t * 16 + m16;
    float bv = bias ? bias[ch] : 0.0f;
#pragma unroll
    for (int r = 0; r < 8; ++r) {
      int orow = i0 + r + 8 * hi;
      float v = acc[r] + bv;
      if (relu) v = fmaxf(v, 0.0f);
      int idx = orow * H_ + ch;
      if (outF)  outF[idx] = v;
      if (outBf) outBf[idx] = (__bf16)v;
    }
  }
}

// ---------------------------------------------------------------------------
// Generic 32x32-tile transpose + f32->bf16: dst[b][c][r] = (bf16) src[b][r][c]
// src: [batches][R][C] f32, dst: [batches][C][R] bf16. grid=(R/32, C/32, nb)
// ---------------------------------------------------------------------------
__global__ void __launch_bounds__(256)
transpose_cvt_kernel(const float* __restrict__ src, __bf16* __restrict__ dst,
                     int R, int C, int srcBS, int dstBS)
{
  __shared__ float tile[32][33];
  const int b = blockIdx.z;
  const int r0 = blockIdx.x * 32, c0 = blockIdx.y * 32;
  const int tx = threadIdx.x, ty = threadIdx.y;            // (32, 8)
  const float* s = src + b * srcBS + (r0) * C + c0;
  for (int r = ty; r < 32; r += 8) tile[r][tx] = s[r * C + tx];
  __syncthreads();
  __bf16* d = dst + b * dstBS + (c0) * R + r0;
  for (int r = ty; r < 32; r += 8) d[r * R + tx] = (__bf16)tile[tx][r];
}

// ---------------------------------------------------------------------------
// s1[row] = h[row,:]·a1 ; s2[row] = h[row,:]·a2   (one wave per row)
// ---------------------------------------------------------------------------
__global__ void __launch_bounds__(256)
scores_kernel(const float* __restrict__ h, const float* __restrict__ a1,
              const float* __restrict__ a2, float* __restrict__ s1,
              float* __restrict__ s2)
{
  __shared__ float a1s[H_], a2s[H_];
  const int tid = threadIdx.x;
  a1s[tid] = a1[tid]; a2s[tid] = a2[tid];
  __syncthreads();
  const int lane = tid & 31, wave = tid >> 5;
  const int row = blockIdx.x * 8 + wave;
  const float* hr = h + row * H_;
  float d1 = 0.0f, d2 = 0.0f;
  for (int c = lane; c < H_; c += 32) { float v = hr[c]; d1 += v * a1s[c]; d2 += v * a2s[c]; }
  for (int off = 16; off; off >>= 1) { d1 += __shfl_xor(d1, off); d2 += __shfl_xor(d2, off); }
  if (lane == 0) { s1[row] = d1; s2[row] = d2; }
}

// ---------------------------------------------------------------------------
// Fused mask -> LeakyReLU -> softmax -> att@h -> ReLU for one (b, 16-row tile).
// Block: 128 threads (4 waves). Wave w owns output channels [64w, 64w+64).
// Dynamic LDS: e[16][1024] f32 | att[16][1024] bf16 | s2[1024] | s1[16]
// ---------------------------------------------------------------------------
__global__ void __launch_bounds__(128)
gat_attention_kernel(const float* __restrict__ s1g, const float* __restrict__ s2g,
                     const float* __restrict__ adj, const __bf16* __restrict__ hT,
                     float* __restrict__ outF, __bf16* __restrict__ outBf)
{
  extern __shared__ char smem[];
  float*  e_sh   = (float*)smem;                                  // 64 KB
  __bf16* att_sh = (__bf16*)(smem + 16 * N_ * 4);                 // 32 KB
  float*  s2_sh  = (float*)(smem + 16 * N_ * 4 + 16 * N_ * 2);    // 4 KB
  float*  s1_sh  = s2_sh + N_;                                    // 64 B

  const int tid  = threadIdx.x;
  const int lane = tid & 31, wave = tid >> 5;
  const int b    = blockIdx.x >> 6;
  const int i0   = (blockIdx.x & 63) * 16;

  for (int j = tid; j < N_; j += 128) s2_sh[j] = s2g[b * N_ + j];
  if (tid < 16) s1_sh[tid] = s1g[b * N_ + i0 + tid];
  __syncthreads();

  // e = mask ? leaky(s1_i + s2_j) : NEG   (float4-vectorized, coalesced adj)
  const float* adjb = adj + (b * N_ + i0) * N_;
  for (int q = tid; q < 16 * (N_ / 4); q += 128) {
    int i  = q >> 8;                 // N_/4 == 256 quads per row
    int j4 = (q & 255) * 4;
    v4f a = *(const v4f*)&adjb[i * N_ + j4];
    float s1v = s1_sh[i];
    v4f o;
#pragma unroll
    for (int u = 0; u < 4; ++u) {
      float v = s1v + s2_sh[j4 + u];
      v = (v >= 0.0f) ? v : ALPHA_ * v;
      o[u] = (a[u] > 0.0f) ? v : NEGV;
    }
    *(v4f*)&e_sh[i * N_ + j4] = o;
  }
  __syncthreads();

  // per-row softmax: wave handles 4 rows, lane-strided + shuffle reductions
  for (int rr = 0; rr < 4; ++rr) {
    int row = wave * 4 + rr;
    float* er = e_sh + row * N_;
    float m = -3.0e38f;
    for (int j = lane; j < N_; j += 32) m = fmaxf(m, er[j]);
    for (int off = 16; off; off >>= 1) m = fmaxf(m, __shfl_xor(m, off));
    float s = 0.0f;
    for (int j = lane; j < N_; j += 32) { float p = __expf(er[j] - m); er[j] = p; s += p; }
    for (int off = 16; off; off >>= 1) s += __shfl_xor(s, off);
    float inv = 1.0f / s;
    __bf16* ar = att_sh + row * N_;
    for (int j = lane; j < N_; j += 32) ar[j] = (__bf16)(er[j] * inv);
  }
  __syncthreads();

  // out[i0+m][c] = relu( sum_j att[m][j] * hT[c][j] ), 128 WMMAs per wave
  const int hi = lane >> 4, m16 = lane & 15;
  v8f acc[4] = {{}, {}, {}, {}};
  const __bf16* hTb = hT + b * H_ * N_;
  for (int kb = 0; kb < N_; kb += 32) {
    v8bf alo = *(const v8bf*)&att_sh[m16 * N_ + kb + hi * 16];
    v8bf ahi = *(const v8bf*)&att_sh[m16 * N_ + kb + 8 + hi * 16];
    v16bf afrag = cat8(alo, ahi);
#pragma unroll
    for (int t = 0; t < 4; ++t) {
      int ch = wave * 64 + t * 16 + m16;
      v16bf bfrag = *(const v16bf*)&hTb[ch * N_ + kb + 16 * hi];
      acc[t] = __builtin_amdgcn_wmma_f32_16x16x32_bf16(false, afrag, false, bfrag,
                                                       (short)0, acc[t], false, false);
    }
  }
#pragma unroll
  for (int t = 0; t < 4; ++t) {
    int ch = wave * 64 + t * 16 + m16;
#pragma unroll
    for (int r = 0; r < 8; ++r) {
      int row = i0 + r + 8 * hi;
      float v = fmaxf(acc[t][r], 0.0f);
      int idx = (b * N_ + row) * H_ + ch;
      if (outF)  outF[idx] = v;
      if (outBf) outBf[idx] = (__bf16)v;
    }
  }
}

// ---------------------------------------------------------------------------
// g0[b][c] = mean_n x[b][n][c] + max_n x[b][n][c]  (bf16 for MLP GEMM input)
// ---------------------------------------------------------------------------
__global__ void __launch_bounds__(256)
pool_kernel(const float* __restrict__ x, __bf16* __restrict__ g0)
{
  const int b = blockIdx.x, ch = threadIdx.x;
  const float* xb = x + b * N_ * H_ + ch;
  float s = 0.0f, m = -3.0e38f;
  for (int n = 0; n < N_; ++n) { float v = xb[n * H_]; s += v; m = fmaxf(m, v); }
  g0[b * H_ + ch] = (__bf16)(s * (1.0f / N_) + m);
}

// ---------------------------------------------------------------------------
extern "C" void kernel_launch(void* const* d_in, const int* in_sizes, int n_in,
                              void* d_out, int out_size, void* d_ws, size_t ws_size,
                              hipStream_t stream)
{
  (void)in_sizes; (void)n_in; (void)out_size; (void)ws_size;
  const float* nf    = (const float*)d_in[0];
  const float* adj   = (const float*)d_in[1];
  const float* emb_W = (const float*)d_in[2];
  const float* emb_b = (const float*)d_in[3];
  const float* W0    = (const float*)d_in[4];
  const float* a1_0  = (const float*)d_in[5];
  const float* a2_0  = (const float*)d_in[6];
  const float* W1    = (const float*)d_in[7];
  const float* a1_1  = (const float*)d_in[8];
  const float* a2_1  = (const float*)d_in[9];
  const float* gpW1  = (const float*)d_in[10];
  const float* gpb1  = (const float*)d_in[11];
  const float* gpW2  = (const float*)d_in[12];
  const float* gpb2  = (const float*)d_in[13];

  float* x_out = (float*)d_out;                     // [B,N,H]
  float* g_out = x_out + (size_t)B_ * N_ * H_;      // [B,H]

  char* ws = (char*)d_ws;
  size_t off = 0;
  auto alloc = [&](size_t bytes) -> char* {
    char* p = ws + off; off += (bytes + 255) & ~(size_t)255; return p;
  };
  __bf16* x0    = (__bf16*)alloc((size_t)B_ * N_ * H_ * 2);
  __bf16* x1    = (__bf16*)alloc((size_t)B_ * N_ * H_ * 2);
  float*  hbuf  = (float*) alloc((size_t)B_ * N_ * H_ * 4);
  __bf16* hT    = (__bf16*)alloc((size_t)B_ * N_ * H_ * 2);
  float*  s1    = (float*) alloc((size_t)B_ * N_ * 4);
  float*  s2    = (float*) alloc((size_t)B_ * N_ * 4);
  __bf16* g0    = (__bf16*)alloc((size_t)B_ * H_ * 2);
  __bf16* tmlp  = (__bf16*)alloc((size_t)B_ * H_ * 2);
  __bf16* embWt = (__bf16*)alloc((size_t)FIN * H_ * 2);   // [H][FIN]
  __bf16* W0t   = (__bf16*)alloc((size_t)H_ * H_ * 2);    // [H][H]
  __bf16* W1t   = (__bf16*)alloc((size_t)H_ * H_ * 2);
  __bf16* gpW1t = (__bf16*)alloc((size_t)H_ * H_ * 2);
  __bf16* gpW2t = (__bf16*)alloc((size_t)H_ * H_ * 2);

  const int M = B_ * N_;
  const size_t att_smem = (size_t)16 * N_ * 4 + (size_t)16 * N_ * 2 + (N_ + 16) * 4;
  dim3 tblk(32, 8);
  dim3 wt_grid64(FIN / 32, H_ / 32, 1);
  dim3 wt_grid256(H_ / 32, H_ / 32, 1);
  dim3 hT_grid(N_ / 32, H_ / 32, B_);

  // one-time weight pre-transposes (f32 [K][H] -> bf16 [H][K])
  transpose_cvt_kernel<<<wt_grid64,  tblk, 0, stream>>>(emb_W, embWt, FIN, H_, 0, 0);
  transpose_cvt_kernel<<<wt_grid256, tblk, 0, stream>>>(W0,   W0t,   H_, H_, 0, 0);
  transpose_cvt_kernel<<<wt_grid256, tblk, 0, stream>>>(W1,   W1t,   H_, H_, 0, 0);
  transpose_cvt_kernel<<<wt_grid256, tblk, 0, stream>>>(gpW1, gpW1t, H_, H_, 0, 0);
  transpose_cvt_kernel<<<wt_grid256, tblk, 0, stream>>>(gpW2, gpW2t, H_, H_, 0, 0);

  // embedding: x0 = nf @ emb_W + emb_b
  gemm_wmma_kernel<false><<<M / 16, 256, 0, stream>>>(nf, embWt, emb_b,
                                                      nullptr, x0, FIN, 0);
  // GAT layer 0
  gemm_wmma_kernel<true><<<M / 16, 256, 0, stream>>>(x0, W0t, nullptr,
                                                     hbuf, nullptr, H_, 0);
  scores_kernel<<<M / 8, 256, 0, stream>>>(hbuf, a1_0, a2_0, s1, s2);
  transpose_cvt_kernel<<<hT_grid, tblk, 0, stream>>>(hbuf, hT, N_, H_,
                                                     N_ * H_, H_ * N_);
  gat_attention_kernel<<<B_ * (N_ / 16), 128, att_smem, stream>>>(
      s1, s2, adj, hT, nullptr, x1);
  // GAT layer 1 (writes final x f32 straight into d_out)
  gemm_wmma_kernel<true><<<M / 16, 256, 0, stream>>>(x1, W1t, nullptr,
                                                     hbuf, nullptr, H_, 0);
  scores_kernel<<<M / 8, 256, 0, stream>>>(hbuf, a1_1, a2_1, s1, s2);
  transpose_cvt_kernel<<<hT_grid, tblk, 0, stream>>>(hbuf, hT, N_, H_,
                                                     N_ * H_, H_ * N_);
  gat_attention_kernel<<<B_ * (N_ / 16), 128, att_smem, stream>>>(
      s1, s2, adj, hT, x_out, nullptr);
  // pooling + 2-layer MLP head
  pool_kernel<<<B_, 256, 0, stream>>>(x_out, g0);
  gemm_wmma_kernel<true><<<1, 256, 0, stream>>>(g0, gpW1t, gpb1,
                                                nullptr, tmlp, H_, 1);
  gemm_wmma_kernel<true><<<1, 256, 0, stream>>>(tmlp, gpW2t, gpb2,
                                                g_out, nullptr, H_, 0);
}
